// OuterProductMean_3934190044005
// MI455X (gfx1250) — compile-verified, matrix operand order checked by reading, not run on hardware
//
#include <hip/hip_runtime.h>

// ---------------------------------------------------------------------------
// OuterProductMean (AlphaFold) for MI455X / gfx1250, wave32 + WMMA bf16 + TDM.
//   S=128 seq, R=384 res, C_M=256, C_H=32, C_Z=128
// Pipeline: prep (weights->bf16) -> lnproj (LN + proj, WMMA) ->
//           transpose (to K-contiguous layouts) ->
//           fused (TDM-staged GEMM1 -> LDS -> GEMM2 -> norm)
// ---------------------------------------------------------------------------

#define S_DIM 128
#define R_DIM 384
#define CM    256
#define CH    32
#define CZ    128
#define NROWS (S_DIM * R_DIM)   // 49152 (s,r) rows
#define MBIG  (R_DIM * CH)      // 12288
#define LN_EPS 1e-5f
#define OPM_EPS 1e-3f

typedef __attribute__((ext_vector_type(16))) __bf16 v16bf;
typedef __attribute__((ext_vector_type(8)))  __bf16 v8bf;
typedef __attribute__((ext_vector_type(4)))  __bf16 v4bf;
typedef __attribute__((ext_vector_type(8)))  float  v8f;
typedef __attribute__((ext_vector_type(4)))  unsigned int v4u;
typedef __attribute__((ext_vector_type(8)))  int    v8i;
typedef __attribute__((ext_vector_type(4)))  int    v4i;

// CDNA5 16-bit A/B operand: lane (l&15) owns row/col, K in two contiguous
// 8-element (16B) runs at (l>>4)*8 and (l>>4)*8 + 16.
static __device__ __forceinline__ v16bf load_frag(const __bf16* p) {
  v8bf lo = *(const v8bf*)(p);
  v8bf hi = *(const v8bf*)(p + 16);
  return __builtin_shufflevector(lo, hi, 0, 1, 2, 3, 4, 5, 6, 7,
                                 8, 9, 10, 11, 12, 13, 14, 15);
}

static __device__ __forceinline__ v8f wmma_bf16(v16bf a, v16bf b, v8f c) {
  // 8 args: (neg_a, A, neg_b, B, c_mod, C, reuse_a, reuse_b)
  return __builtin_amdgcn_wmma_f32_16x16x32_bf16(false, a, false, b,
                                                 (short)0, c, false, false);
}

// ---------------------------------------------------------------------------
// TDM: DMA one contiguous 256-row x 256-byte tile (64 KB) from global into
// LDS, inserting 16B of padding after every 256B row (pitch 272B) so the
// subsequent ds_load_b128 fragment reads are bank-conflict-free.
// D# per CDNA5 ISA ch.8: group0 = {ctl, lds_addr, gaddr_lo, gaddr_hi|type},
// group1 = {flags, dims/tiles/strides}, groups 2/3 unused (<=2D tensor).
// data_size = 8B (code 3): tile_dim0=32, tile_dim1=256, stride0=32.
// pad_interval code 5 (=64 dwords = one row), pad_amount code 3 (=4 dwords).
// Toolchain builtin is the 6-arg clang-23 form:
//   (uint32x4 g0, int32x8 g1, int32x4 g2, int32x4 g3, int32x8 g4, i32 cpol)
// ---------------------------------------------------------------------------
static __device__ __forceinline__ void tdm_load_tile64k(unsigned lds_addr,
                                                        const void* gptr) {
  unsigned long long ga = (unsigned long long)gptr;
  v4u g0;
  g0[0] = 1u;                                        // count=1, user mode
  g0[1] = lds_addr;                                  // LDS dest (bytes)
  g0[2] = (unsigned)(ga & 0xffffffffu);              // global_addr[31:0]
  g0[3] = (unsigned)((ga >> 32) & 0x1ffffffu)        // global_addr[56:32]
        | (2u << 30);                                // type=2 ("image")
  v8i g1;
  g1[0] = (int)((3u << 16)                           // data_size = 8B
              | (1u << 20)                           // pad_enable
              | (5u << 22)                           // pad_interval: 64 dw
              | (3u << 25));                         // pad_amount: 4 dw
  g1[1] = (int)(32u << 16);                          // tensor_dim0 = 32
  g1[2] = (int)(256u << 16);                         // tensor_dim1 = 256
  g1[3] = (int)(32u << 16);                          // tile_dim0 = 32
  g1[4] = (int)(256u);                               // tile_dim1=256, dim2=0
  g1[5] = 32;                                        // tensor_dim0_stride
  g1[6] = 0;
  g1[7] = 0;
  v4i gz4 = v4i{};
  v8i gz8 = v8i{};
  __builtin_amdgcn_tensor_load_to_lds(g0, g1, gz4, gz4, gz8, 0);
}

// ---------------------------------------------------------------------------
// Kernel 0: weights -> bf16.  W12[64][256] = {w1;w2} (B-operand, row=n, k cont)
//           Wout[128][1024] = w_out (B-operand for stage 2)
// ---------------------------------------------------------------------------
__global__ __launch_bounds__(256) void opm_prep(
    const float* __restrict__ w1, const float* __restrict__ w2,
    const float* __restrict__ w_out,
    __bf16* __restrict__ W12, __bf16* __restrict__ Wout) {
  int idx = blockIdx.x * 256 + threadIdx.x;
  if (idx < 64 * 256) {
    int n = idx >> 8, k = idx & 255;
    float v = (n < 32) ? w1[n * 256 + k] : w2[(n - 32) * 256 + k];
    W12[idx] = (__bf16)v;
  }
  if (idx < 128 * 1024) {
    Wout[idx] = (__bf16)w_out[idx];
  }
}

// ---------------------------------------------------------------------------
// Kernel 1: LayerNorm + projections. 64 rows (p = s*384+r) per WG, 8 waves.
// Phase 1: fp32 LN (wave32 shuffle reduction) -> bf16 LDS tile [64][272].
// Phase 2: WMMA  [64 x 256] x [256 x 64] -> a,b with bias+mask.
// Output layout: aNat/bNat flat index p*32 + h  (coalesced bf16 stores).
// ---------------------------------------------------------------------------
__global__ __launch_bounds__(256) void opm_lnproj(
    const float* __restrict__ m, const float* __restrict__ mask,
    const float* __restrict__ ln_w, const float* __restrict__ ln_b,
    const float* __restrict__ b1, const float* __restrict__ b2,
    const __bf16* __restrict__ W12,
    __bf16* __restrict__ aNat, __bf16* __restrict__ bNat) {
  __shared__ __bf16 lnT[64][272];  // 256 + 16 pad -> bank-staggered rows
  const int tid = threadIdx.x;
  const int wid = tid >> 5, lane = tid & 31;
  const int hi = lane >> 4, ln15 = lane & 15;
  const long p0 = (long)blockIdx.x * 64;

  // Per-lane LN params for channels [lane*8, lane*8+8)
  float4 wl = *(const float4*)(ln_w + lane * 8);
  float4 wh = *(const float4*)(ln_w + lane * 8 + 4);
  float4 bl = *(const float4*)(ln_b + lane * 8);
  float4 bh = *(const float4*)(ln_b + lane * 8 + 4);

  for (int rl = 0; rl < 8; ++rl) {
    int row = wid * 8 + rl;
    long p = p0 + row;
    const float* mp = m + p * CM + lane * 8;
    float4 x0 = *(const float4*)(mp);
    float4 x1 = *(const float4*)(mp + 4);
    float s  = x0.x + x0.y + x0.z + x0.w + x1.x + x1.y + x1.z + x1.w;
    float s2 = x0.x * x0.x + x0.y * x0.y + x0.z * x0.z + x0.w * x0.w +
               x1.x * x1.x + x1.y * x1.y + x1.z * x1.z + x1.w * x1.w;
    #pragma unroll
    for (int off = 16; off > 0; off >>= 1) {
      s  += __shfl_xor(s, off, 32);
      s2 += __shfl_xor(s2, off, 32);
    }
    float mu  = s * (1.0f / 256.0f);
    float var = s2 * (1.0f / 256.0f) - mu * mu;
    float rs  = rsqrtf(var + LN_EPS);
    __bf16* dst = &lnT[row][lane * 8];
    dst[0] = (__bf16)((x0.x - mu) * rs * wl.x + bl.x);
    dst[1] = (__bf16)((x0.y - mu) * rs * wl.y + bl.y);
    dst[2] = (__bf16)((x0.z - mu) * rs * wl.z + bl.z);
    dst[3] = (__bf16)((x0.w - mu) * rs * wl.w + bl.w);
    dst[4] = (__bf16)((x1.x - mu) * rs * wh.x + bh.x);
    dst[5] = (__bf16)((x1.y - mu) * rs * wh.y + bh.y);
    dst[6] = (__bf16)((x1.z - mu) * rs * wh.z + bh.z);
    dst[7] = (__bf16)((x1.w - mu) * rs * wh.w + bh.w);
  }
  __syncthreads();

  // Phase 2: 16 output tiles (4 m x 4 n), 2 per wave.
  const int mtile = wid & 3;
  const int nt0 = (wid >> 2) * 2;
  v8f acc0 = v8f{}, acc1 = v8f{};
  #pragma unroll
  for (int kk = 0; kk < 8; ++kk) {
    int k0 = kk * 32 + hi * 8;
    v16bf af  = load_frag(&lnT[mtile * 16 + ln15][k0]);
    v16bf bf0 = load_frag(W12 + (long)((nt0 + 0) * 16 + ln15) * 256 + k0);
    v16bf bf1 = load_frag(W12 + (long)((nt0 + 1) * 16 + ln15) * 256 + k0);
    acc0 = wmma_bf16(af, bf0, acc0);
    acc1 = wmma_bf16(af, bf1, acc1);
  }
  #pragma unroll
  for (int t = 0; t < 2; ++t) {
    int ng = (nt0 + t) * 16 + ln15;               // 0..63
    float bias = (ng < 32) ? b1[ng] : b2[ng - 32];
    v8f acc = t ? acc1 : acc0;
    __bf16* dst = (ng < 32) ? aNat : bNat;
    int h = (ng < 32) ? ng : ng - 32;
    #pragma unroll
    for (int v = 0; v < 8; ++v) {
      int rowL = mtile * 16 + v + 8 * hi;         // C layout: lane hi half +8
      long p = p0 + rowL;
      float val = (acc[v] + bias) * mask[p];      // mask[s][r] flat == mask[p]
      dst[p * 32 + h] = (__bf16)val;
    }
  }
}

// ---------------------------------------------------------------------------
// Kernel 2: transpose [S=128][MBIG=12288] -> [MBIG][128] for both a and b,
// so GEMM-1 operand fragments are K(=s)-contiguous. 32x32 LDS tiles.
// ---------------------------------------------------------------------------
__global__ __launch_bounds__(256) void opm_transpose(
    const __bf16* __restrict__ aNat, const __bf16* __restrict__ bNat,
    __bf16* __restrict__ Abig, __bf16* __restrict__ Bbig) {
  __shared__ __bf16 tile[32][36];
  const __bf16* in = blockIdx.z ? bNat : aNat;
  __bf16* out = blockIdx.z ? Bbig : Abig;
  const int m0 = blockIdx.x * 32, s0 = blockIdx.y * 32;
  const int tr = threadIdx.x >> 3;
  const int tc = (threadIdx.x & 7) * 4;
  v4bf v = *(const v4bf*)(in + (long)(s0 + tr) * MBIG + m0 + tc);
  tile[tr][tc + 0] = v[0];
  tile[tr][tc + 1] = v[1];
  tile[tr][tc + 2] = v[2];
  tile[tr][tc + 3] = v[3];
  __syncthreads();
  v4bf o;
  o[0] = tile[tc + 0][tr];
  o[1] = tile[tc + 1][tr];
  o[2] = tile[tc + 2][tr];
  o[3] = tile[tc + 3][tr];
  *(v4bf*)(out + (long)(m0 + tr) * S_DIM + s0 + tc) = o;
}

// ---------------------------------------------------------------------------
// Kernel 3: fused GEMM1 + GEMM2 + normalization.
// WG = 512 threads (16 waves), 8x8 (i,j)-pair tile.
//  Stage 0: TDM-stage A tile (64 KB) and B tile (64 KB) into LDS with
//           DMA-inserted row padding (272 B pitch); s_wait_tensorcnt.
//  Stage 1: O[256][256] = A-tile x B-tile over K=128 (4x4 WMMA tiles/wave),
//           operands read from LDS only.  O -> bf16 -> LDS (stage-2 A
//           operand [pair=64][k=c*32+e]), overlaying the dead staging region.
//  Stage 2: out2[64][128] = O x Wout^T over K=1024, + b_out, * 1/(norm+eps).
// Dynamic LDS: max(staging 2*69632, A2 64*1032*2) + invn = 139520 B (<320 KB)
// ---------------------------------------------------------------------------
#define A2_PITCH 1032      // 1024 + 8 pad: 2064B rows -> bank-staggered
#define STG_PITCH 136      // 272 B rows (256 data + 16 DMA pad)
#define STG_BYTES (256 * 272)

__global__ __launch_bounds__(512) void opm_fused(
    const __bf16* __restrict__ Abig, const __bf16* __restrict__ Bbig,
    const __bf16* __restrict__ Wout, const float* __restrict__ mask,
    const float* __restrict__ b_out, float* __restrict__ out) {
  extern __shared__ char smem[];
  __bf16* As = (__bf16*)smem;                        // [256][STG_PITCH]
  __bf16* Bs = (__bf16*)(smem + STG_BYTES);          // [256][STG_PITCH]
  __bf16* A2 = (__bf16*)smem;                        // overlays staging later
  float* invn = (float*)(smem + 2 * STG_BYTES);      // [64]

  const int tid = threadIdx.x;
  const int wid = tid >> 5, lane = tid & 31;
  const int hi = lane >> 4, ln15 = lane & 15;
  const int i0 = blockIdx.x * 8, j0 = blockIdx.y * 8;

  // ---- Stage 0: issue the two tile DMAs (one wave each; TDM is per-wave) --
  if (wid == 0) {
    unsigned ldsA = (unsigned)(unsigned long long)(void*)As;
    tdm_load_tile64k(ldsA, Abig + (long)i0 * 32 * S_DIM);
  }
  if (wid == 1) {
    unsigned ldsB = (unsigned)(unsigned long long)(void*)Bs;
    tdm_load_tile64k(ldsB, Bbig + (long)j0 * 32 * S_DIM);
  }

  // Overlap: 1/(sum_s mask[s][i]*mask[s][j] + eps) for the 64 pairs.
  if (tid < 64) {
    int di = tid >> 3, dj = tid & 7;
    const float* mi = mask + (i0 + di);
    const float* mj = mask + (j0 + dj);
    float sum = 0.0f;
    for (int s = 0; s < S_DIM; ++s) sum += mi[s * R_DIM] * mj[s * R_DIM];
    invn[tid] = 1.0f / (sum + OPM_EPS);
  }

  __builtin_amdgcn_s_wait_tensorcnt(0);  // NOP for waves with TENSORcnt==0
  __syncthreads();

  // ---- Stage 1: 256x256x128 bf16 GEMM; wave (wm,wn) owns 64x64 subtile ----
  const int wm = wid >> 2, wn = wid & 3;
  v8f acc[4][4];
  #pragma unroll
  for (int a = 0; a < 4; ++a)
    #pragma unroll
    for (int b = 0; b < 4; ++b)
      acc[a][b] = v8f{};

  #pragma unroll
  for (int kk = 0; kk < 4; ++kk) {
    int k0 = kk * 32 + hi * 8;
    v16bf af[4], bf[4];
    #pragma unroll
    for (int t = 0; t < 4; ++t) {
      af[t] = load_frag(As + (wm * 64 + t * 16 + ln15) * STG_PITCH + k0);
      bf[t] = load_frag(Bs + (wn * 64 + t * 16 + ln15) * STG_PITCH + k0);
    }
    #pragma unroll
    for (int tm = 0; tm < 4; ++tm)
      #pragma unroll
      for (int tn = 0; tn < 4; ++tn)
        acc[tm][tn] = wmma_bf16(af[tm], bf[tn], acc[tm][tn]);
  }

  // Staging region is dead once every wave is done reading it.
  __syncthreads();

  // O (f32 accums) -> bf16 LDS in stage-2 A-operand layout [pair][c*32+e].
  #pragma unroll
  for (int tm = 0; tm < 4; ++tm) {
    #pragma unroll
    for (int tn = 0; tn < 4; ++tn) {
      int n = wn * 64 + tn * 16 + ln15;
      int dj = n >> 5, e = n & 31;
      #pragma unroll
      for (int v = 0; v < 8; ++v) {
        int mrow = wm * 64 + tm * 16 + v + 8 * hi;
        int di = mrow >> 5, c = mrow & 31;
        A2[(di * 8 + dj) * A2_PITCH + c * 32 + e] = (__bf16)acc[tm][tn][v];
      }
    }
  }
  __syncthreads();

  // ---- Stage 2: [64 pairs] x [K=1024] x [Z=128]; 2 tiles per wave ----
  const int tn2 = wid & 7;           // z-tile
  const int mt0 = (wid >> 3) * 2;    // first of two pair-tiles
  const int zrow = tn2 * 16 + ln15;
  v8f acc2[2];
  acc2[0] = v8f{};
  acc2[1] = v8f{};
  #pragma unroll 4
  for (int kk = 0; kk < 32; ++kk) {
    int k0 = kk * 32 + hi * 8;
    v16bf bfz = load_frag(Wout + (long)zrow * 1024 + k0);
    #pragma unroll
    for (int t = 0; t < 2; ++t) {
      v16bf afp = load_frag(A2 + (long)((mt0 + t) * 16 + ln15) * A2_PITCH + k0);
      acc2[t] = wmma_bf16(afp, bfz, acc2[t]);
    }
  }

  float bz = b_out[zrow];
  #pragma unroll
  for (int t = 0; t < 2; ++t) {
    #pragma unroll
    for (int v = 0; v < 8; ++v) {
      int pr = (mt0 + t) * 16 + v + 8 * hi;  // pair index 0..63
      int di = pr >> 3, dj = pr & 7;
      long o = ((long)(i0 + di) * R_DIM + (j0 + dj)) * CZ + zrow;
      out[o] = (acc2[t][v] + bz) * invn[pr];
    }
  }
}

// ---------------------------------------------------------------------------
// Host-side launcher
// ---------------------------------------------------------------------------
extern "C" void kernel_launch(void* const* d_in, const int* in_sizes, int n_in,
                              void* d_out, int out_size, void* d_ws, size_t ws_size,
                              hipStream_t stream) {
  const float* m      = (const float*)d_in[0];
  const float* mask   = (const float*)d_in[1];
  const float* ln_w   = (const float*)d_in[2];
  const float* ln_b   = (const float*)d_in[3];
  const float* w1     = (const float*)d_in[4];
  const float* b1     = (const float*)d_in[5];
  const float* w2     = (const float*)d_in[6];
  const float* b2     = (const float*)d_in[7];
  const float* w_out  = (const float*)d_in[8];
  const float* b_out  = (const float*)d_in[9];
  float* out = (float*)d_out;

  char* ws = (char*)d_ws;
  const size_t OFF_W12  = 0;                       // 64*256*2      = 32 KB
  const size_t OFF_WOUT = 32768;                   // 128*1024*2    = 256 KB
  const size_t OFF_ANAT = 294912;                  // 49152*32*2    = 3 MB
  const size_t OFF_BNAT = OFF_ANAT + 3145728;
  const size_t OFF_ABIG = OFF_BNAT + 3145728;
  const size_t OFF_BBIG = OFF_ABIG + 3145728;      // end: ~12.9 MB
  __bf16* W12  = (__bf16*)(ws + OFF_W12);
  __bf16* Wout = (__bf16*)(ws + OFF_WOUT);
  __bf16* aNat = (__bf16*)(ws + OFF_ANAT);
  __bf16* bNat = (__bf16*)(ws + OFF_BNAT);
  __bf16* Abig = (__bf16*)(ws + OFF_ABIG);
  __bf16* Bbig = (__bf16*)(ws + OFF_BBIG);

  // 0) weights -> bf16
  opm_prep<<<512, 256, 0, stream>>>(w1, w2, w_out, W12, Wout);
  // 1) LN + projections (WMMA), natural layout
  opm_lnproj<<<NROWS / 64, 256, 0, stream>>>(m, mask, ln_w, ln_b, b1, b2,
                                             W12, aNat, bNat);
  // 2) transpose to K-contiguous GEMM layouts
  opm_transpose<<<dim3(MBIG / 32, S_DIM / 32, 2), 256, 0, stream>>>(
      aNat, bNat, Abig, Bbig);
  // 3) fused outer-product GEMM + projection + normalization
  const size_t smem = 2 * STG_BYTES + 256;         // 139520 B
  opm_fused<<<dim3(R_DIM / 8, R_DIM / 8), 512, smem, stream>>>(
      Abig, Bbig, Wout, mask, b_out, out);
}
